// TemporalFusionModule_22806276342086
// MI455X (gfx1250) — matrix operand order repaired and checked by reading, not compiled, not run on hardware
//
#include <hip/hip_runtime.h>

// ---------------------------------------------------------------------------
// Mamba block, last-token output, for MI455X (gfx1250, wave32, WMMA).
//   B=4, L=2048, d_model=512, d_inner=1024, d_state=16, d_conv=4, dt_rank=32
// ---------------------------------------------------------------------------

typedef __attribute__((ext_vector_type(16))) _Float16 v16h;
typedef __attribute__((ext_vector_type(8)))  float    v8f;

constexpr int Bb = 4;
constexpr int Ls = 2048;
constexpr int DM = 512;    // d_model
constexpr int DS = 16;     // d_state
constexpr int DI = 1024;   // d_inner
constexpr int DR = 32;     // dt_rank
constexpr int MROWS = Bb * Ls;  // 8192 token rows

#define WMMA_F16(A, B, C) \
  __builtin_amdgcn_wmma_f32_16x16x32_f16(false, (A), false, (B), (short)0, (C), false, false)

// ---------------------------------------------------------------------------
// WMMA fragment loaders (wave32, contiguous k = 16*(lane/16) + s in both A and B), fp32 -> f16.
// ---------------------------------------------------------------------------

// A fragment: 16x32 (MxK) tile from row-major A[M x lda].
__device__ __forceinline__ v16h load_a_frag(const float* __restrict__ A, int lda,
                                            int m_base, int k_base, int lane) {
  int r  = m_base + (lane & 15);
  int kb = k_base + ((lane >> 4) << 4);           // 0 or 16
  const float* p = A + (size_t)r * lda + kb;
  v16h f;
#pragma unroll
  for (int i = 0; i < 8; ++i) f[i]     = (_Float16)p[i];        // K = kb+i
#pragma unroll
  for (int i = 0; i < 8; ++i) f[8 + i] = (_Float16)p[8 + i];    // K = kb+8+i
  return f;
}

// B fragment: 32x16 (KxN) with B[k][n] = W[n][k], W row-major [N x ldw].
__device__ __forceinline__ v16h load_b_frag_wt(const float* __restrict__ W, int ldw,
                                               int n_base, int k_base, int lane) {
  int n  = n_base + (lane & 15);
  int kb = k_base + ((lane >> 4) << 4);           // 0 or 16
  const float* p = W + (size_t)n * ldw + kb;
  v16h f;
#pragma unroll
  for (int i = 0; i < 16; ++i) f[i] = (_Float16)p[i];
  return f;
}

__device__ __forceinline__ void store_tile(float* __restrict__ C, int ldc,
                                           int m_base, int n_base, int lane,
                                           const v8f& acc) {
  int n  = n_base + (lane & 15);
  int m0 = m_base + ((lane >> 4) << 3);
#pragma unroll
  for (int j = 0; j < 8; ++j) C[(size_t)(m0 + j) * ldc + n] = acc[j];
}

// ---------------------------------------------------------------------------
// Kernel 1: x-half of in_proj.  X[8192,1024] = x_seq[8192,512] @ Wx[1024,512]^T
// Each wave computes a 16x64 strip: one A fragment feeds 4 WMMAs per K-step.
// ---------------------------------------------------------------------------
__global__ void k_inproj_x(const float* __restrict__ Xin,
                           const float* __restrict__ Win,
                           float* __restrict__ Xout) {
  int lane  = threadIdx.x & 31;
  int strip = blockIdx.x * (blockDim.x >> 5) + (threadIdx.x >> 5); // 8192 strips
  int tn = (strip & 15) << 6;   // 16 N-strips of width 64
  int tm = (strip >> 4) << 4;   // 512 M tiles
  v8f a0 = {}, a1 = {}, a2 = {}, a3 = {};
  for (int k = 0; k < DM; k += 32) {
    v16h a  = load_a_frag(Xin, DM, tm, k, lane);
    v16h b0 = load_b_frag_wt(Win, DM, tn +  0, k, lane);
    v16h b1 = load_b_frag_wt(Win, DM, tn + 16, k, lane);
    v16h b2 = load_b_frag_wt(Win, DM, tn + 32, k, lane);
    v16h b3 = load_b_frag_wt(Win, DM, tn + 48, k, lane);
    a0 = WMMA_F16(a, b0, a0);
    a1 = WMMA_F16(a, b1, a1);
    a2 = WMMA_F16(a, b2, a2);
    a3 = WMMA_F16(a, b3, a3);
  }
  store_tile(Xout, DI, tm, tn +  0, lane, a0);
  store_tile(Xout, DI, tm, tn + 16, lane, a1);
  store_tile(Xout, DI, tm, tn + 32, lane, a2);
  store_tile(Xout, DI, tm, tn + 48, lane, a3);
}

// ---------------------------------------------------------------------------
// Kernel 2: z-half of in_proj, LAST TOKEN ONLY. z_last[4,1024].
// ---------------------------------------------------------------------------
__global__ void k_z_last(const float* __restrict__ Xin,
                         const float* __restrict__ Win,
                         float* __restrict__ zlast) {
  int idx = blockIdx.x * 256 + threadIdx.x;     // 4096 = 4*1024
  int b = idx >> 10, n = idx & 1023;
  const float* xr = Xin + ((size_t)b * Ls + (Ls - 1)) * DM;
  const float* w  = Win + (size_t)(DI + n) * DM;
  float acc = 0.f;
#pragma unroll 4
  for (int k = 0; k < DM; ++k) acc += xr[k] * w[k];
  zlast[idx] = acc;
}

// ---------------------------------------------------------------------------
// Kernel 3: causal depthwise conv (d_conv=4) + bias + SiLU, IN PLACE.
// ---------------------------------------------------------------------------
__global__ void k_conv_silu(float* __restrict__ X,
                            const float* __restrict__ cw,
                            const float* __restrict__ cb) {
  int idx = blockIdx.x * 256 + threadIdx.x;     // 4096
  int b = idx >> 10, d = idx & 1023;
  float w0 = cw[d * 4 + 0], w1 = cw[d * 4 + 1];
  float w2 = cw[d * 4 + 2], w3 = cw[d * 4 + 3];
  float bias = cb[d];
  float p1 = 0.f, p2 = 0.f, p3 = 0.f;           // x[t-1], x[t-2], x[t-3]
  size_t base = (size_t)b * Ls * DI + d;
  for (int t = 0; t < Ls; ++t) {
    float cur = X[base + (size_t)t * DI];
    float v = bias + w0 * p3 + w1 * p2 + w2 * p1 + w3 * cur;
    X[base + (size_t)t * DI] = v / (1.f + __expf(-v));   // SiLU
    p3 = p2; p2 = p1; p1 = cur;
  }
}

// ---------------------------------------------------------------------------
// Kernel 4: x_proj.  x_dbl[8192,64] = xc[8192,1024] @ Wxp[64,1024]^T
// One wave per 16-row strip produces all 64 output columns (4 WMMAs/K-step).
// ---------------------------------------------------------------------------
__global__ void k_xproj(const float* __restrict__ XC,
                        const float* __restrict__ W,
                        float* __restrict__ xdbl) {
  int lane = threadIdx.x & 31;
  int tm   = (blockIdx.x * (blockDim.x >> 5) + (threadIdx.x >> 5)) << 4; // 512 strips
  v8f a0 = {}, a1 = {}, a2 = {}, a3 = {};
  for (int k = 0; k < DI; k += 32) {
    v16h a  = load_a_frag(XC, DI, tm, k, lane);
    v16h b0 = load_b_frag_wt(W, DI,  0, k, lane);
    v16h b1 = load_b_frag_wt(W, DI, 16, k, lane);
    v16h b2 = load_b_frag_wt(W, DI, 32, k, lane);
    v16h b3 = load_b_frag_wt(W, DI, 48, k, lane);
    a0 = WMMA_F16(a, b0, a0);
    a1 = WMMA_F16(a, b1, a1);
    a2 = WMMA_F16(a, b2, a2);
    a3 = WMMA_F16(a, b3, a3);
  }
  store_tile(xdbl, 64, tm,  0, lane, a0);
  store_tile(xdbl, 64, tm, 16, lane, a1);
  store_tile(xdbl, 64, tm, 32, lane, a2);
  store_tile(xdbl, 64, tm, 48, lane, a3);
}

// ---------------------------------------------------------------------------
// Kernel 5: dt_proj + softplus.  delta[8192,1024] =
//   softplus(x_dbl[:, 0:32] @ Wdt[1024,32]^T + b)   -- K=32: ONE WMMA per tile
// ---------------------------------------------------------------------------
__global__ void k_delta(const float* __restrict__ xdbl,
                        const float* __restrict__ Wdt,
                        const float* __restrict__ bdt,
                        float* __restrict__ delta) {
  int lane = threadIdx.x & 31;
  int tile = blockIdx.x * (blockDim.x >> 5) + (threadIdx.x >> 5);
  int tn = tile & 63;          // 64 N-tiles
  int tm = tile >> 6;          // 512 M-tiles
  v8f acc = {};
  v16h a = load_a_frag(xdbl, 64, tm * 16, 0, lane);     // dt part: cols 0..31
  v16h b = load_b_frag_wt(Wdt, DR, tn * 16, 0, lane);
  acc = WMMA_F16(a, b, acc);
  int n  = tn * 16 + (lane & 15);
  int m0 = tm * 16 + ((lane >> 4) << 3);
  float bias = bdt[n];
#pragma unroll
  for (int j = 0; j < 8; ++j) {
    float v = acc[j] + bias;
    v = (v > 20.f) ? v : log1pf(__expf(v));             // softplus
    delta[(size_t)(m0 + j) * DI + n] = v;
  }
}

// ---------------------------------------------------------------------------
// Kernel 6: selective scan. 16 blocks x 256 threads: one thread per (b,d).
// h[16] in VGPRs. B_t/C_t (32 floats) staged into LDS each step by wave 0
// via the CDNA5 async-to-LDS path (ASYNCcnt-tracked), then published by the
// workgroup barrier. y emitted only at t = L-1.
// ---------------------------------------------------------------------------
__global__ void k_scan(const float* __restrict__ delta,
                       const float* __restrict__ xc,
                       const float* __restrict__ xdbl,
                       const float* __restrict__ A_log,
                       float* __restrict__ yscan) {
  int b = blockIdx.x >> 2;
  int d = ((blockIdx.x & 3) << 8) + threadIdx.x;
  float a[DS], h[DS];
#pragma unroll
  for (int s = 0; s < DS; ++s) { a[s] = -__expf(A_log[d * DS + s]); h[s] = 0.f; }
  __shared__ float sbc[2 * DS];                 // B_t then C_t
  // Raw LDS byte offset: generic pointers to LDS carry the aperture in the
  // high 32 bits, so truncation yields the LDS offset the async op needs.
  unsigned ldsoff = (unsigned)(uintptr_t)(&sbc[threadIdx.x & 31]);
  const size_t rowbase = (size_t)b * Ls;
  for (int t = 0; t < Ls; ++t) {
    size_t r = rowbase + t;
    if (threadIdx.x < 2 * DS) {
      // lane i copies x_dbl[r, 32+i] -> sbc[i]   (GLOBAL_LOAD_ASYNC_TO_LDS_B32)
      const float* gp = xdbl + r * 64 + DR + threadIdx.x;
      asm volatile("global_load_async_to_lds_b32 %0, %1, off"
                   :: "v"(ldsoff), "v"((unsigned long long)(uintptr_t)gp)
                   : "memory");
      asm volatile("s_wait_asynccnt 0x0" ::: "memory");
    }
    __syncthreads();
    float dv = delta[r * DI + d];
    float u  = xc[r * DI + d];
    float du = dv * u;
#pragma unroll
    for (int s = 0; s < DS; ++s)
      h[s] = __expf(dv * a[s]) * h[s] + du * sbc[s];
    if (t == Ls - 1) {
      float y = 0.f;
#pragma unroll
      for (int s = 0; s < DS; ++s) y += h[s] * sbc[DS + s];
      yscan[(size_t)b * DI + d] = y;
    }
    __syncthreads();
  }
}

// ---------------------------------------------------------------------------
// Kernel 7: last-token epilogue: y = (yscan + xc_last*D) * silu(z_last),
// then out[b, 0:512] = y @ Wout^T.  One block per batch, y in LDS.
// ---------------------------------------------------------------------------
__global__ void k_final(const float* __restrict__ yscan,
                        const float* __restrict__ xc,
                        const float* __restrict__ zlast,
                        const float* __restrict__ Dvec,
                        const float* __restrict__ Wout,
                        float* __restrict__ out) {
  int b = blockIdx.x;
  int tid = threadIdx.x;
  __shared__ float ys[DI];
  const size_t last = ((size_t)b * Ls + (Ls - 1)) * DI;
#pragma unroll
  for (int j = 0; j < 4; ++j) {
    int d = tid + j * 256;
    float y = yscan[b * DI + d] + xc[last + d] * Dvec[d];
    float z = zlast[b * DI + d];
    y *= z / (1.f + __expf(-z));                // * silu(z)
    ys[d] = y;
  }
  __syncthreads();
#pragma unroll
  for (int j = 0; j < 2; ++j) {
    int n = tid + j * 256;
    const float* w = Wout + (size_t)n * DI;
    float acc = 0.f;
#pragma unroll 4
    for (int dd = 0; dd < DI; ++dd) acc += ys[dd] * w[dd];
    out[b * DM + n] = acc;
  }
}

// ---------------------------------------------------------------------------
extern "C" void kernel_launch(void* const* d_in, const int* in_sizes, int n_in,
                              void* d_out, int out_size, void* d_ws, size_t ws_size,
                              hipStream_t stream) {
  const float* x_seq     = (const float*)d_in[0];
  const float* in_proj_w = (const float*)d_in[1];
  const float* conv_w    = (const float*)d_in[2];
  const float* conv_b    = (const float*)d_in[3];
  const float* x_proj_w  = (const float*)d_in[4];
  const float* dt_proj_w = (const float*)d_in[5];
  const float* dt_proj_b = (const float*)d_in[6];
  const float* A_log     = (const float*)d_in[7];
  const float* Dvec      = (const float*)d_in[8];
  const float* out_proj_w= (const float*)d_in[9];
  float* out = (float*)d_out;

  // Workspace layout (floats): ~69.2 MB total
  float* ws    = (float*)d_ws;
  float* X     = ws;                          // [8192,1024] x-half -> xc (in place)
  float* delta = ws + (size_t)MROWS * DI;     // [8192,1024]
  float* xdbl  = delta + (size_t)MROWS * DI;  // [8192,64]
  float* zlast = xdbl + (size_t)MROWS * 64;   // [4,1024]
  float* yscan = zlast + Bb * DI;             // [4,1024]

  // 1) in_proj x-half: 512 M-tiles x 16 N-strips = 8192 waves -> 1024 blocks
  k_inproj_x<<<1024, 256, 0, stream>>>(x_seq, in_proj_w, X);
  // 2) z-half, last token only
  k_z_last<<<16, 256, 0, stream>>>(x_seq, in_proj_w, zlast);
  // 3) causal conv + SiLU in place
  k_conv_silu<<<16, 256, 0, stream>>>(X, conv_w, conv_b);
  // 4) x_proj: 512 strips -> 64 blocks
  k_xproj<<<64, 256, 0, stream>>>(X, x_proj_w, xdbl);
  // 5) dt_proj + softplus: 32768 tiles -> 4096 blocks
  k_delta<<<4096, 256, 0, stream>>>(xdbl, dt_proj_w, dt_proj_b, delta);
  // 6) selective scan (async-to-LDS staging of B_t/C_t)
  k_scan<<<16, 256, 0, stream>>>(delta, X, xdbl, A_log, yscan);
  // 7) gate + out_proj, last token
  k_final<<<Bb, 256, 0, stream>>>(yscan, X, zlast, Dvec, out_proj_w, out);
}
